// MultiHeadLatentAttention_26525718020264
// MI455X (gfx1250) — compile-verified, matrix operand order checked by reading, not run on hardware
//
#include <hip/hip_runtime.h>

typedef __bf16 bf16_t;
typedef __attribute__((ext_vector_type(8)))  __bf16 v8bf;
typedef __attribute__((ext_vector_type(16))) __bf16 v16bf;
typedef __attribute__((ext_vector_type(8)))  float  v8f;

#define D_MODEL  2048
#define N_HEADS  16
#define D_HEAD   128
#define D_LATENT 512
#define SEQ      2048
#define BATCH    2
#define MTOT     (BATCH * SEQ)   // 4096 tokens

// Load a 16-element bf16 operand fragment in the CDNA5 WMMA A/B per-lane
// pattern: elements [0..7] = K = +0..7, elements [8..15] = K = +16..23.
// Caller passes (row_ptr + ko) where ko = 8*(lane>=16).
__device__ __forceinline__ v16bf ld_op(const bf16_t* p) {
    v8bf lo = *(const v8bf*)(p);
    v8bf hi = *(const v8bf*)(p + 16);
    return __builtin_shufflevector(lo, hi,
        0,1,2,3,4,5,6,7,8,9,10,11,12,13,14,15);
}

__device__ __forceinline__ v8f wmma_bf16(v16bf a, v16bf b, v8f c) {
    return __builtin_amdgcn_wmma_f32_16x16x32_bf16(
        false, a, false, b, (short)0, c, false, false);
}

// ---------------- elementwise f32 -> bf16 ----------------
__global__ void cvt_kernel(const float* __restrict__ in,
                           bf16_t* __restrict__ out, int n) {
    int i = blockIdx.x * blockDim.x + threadIdx.x;
    if (i < n) out[i] = (bf16_t)in[i];
}

// -------- transpose + convert: in f32 [K,N] -> out bf16 [N,K] --------
__global__ void transpose_cvt_kernel(const float* __restrict__ in,
                                     bf16_t* __restrict__ out,
                                     int K, int N) {
    int i = blockIdx.x * blockDim.x + threadIdx.x;   // i = n*K + k
    if (i >= K * N) return;
    int k = i % K;
    int n = i / K;
    out[i] = (bf16_t)in[(size_t)k * N + n];
}

// ---------------- generic WMMA GEMM ----------------
// C[M,N] = A[M,K] @ B[K,N], with B supplied transposed as BT[N,K] (bf16).
// One wave (32 threads) per 32x64 output tile: 2x4 grid of 16x16 WMMA tiles.
// Epilogue: optional bf16 row-major (Cb), fp32 row-major (Cf), and bf16
// batch-transposed store CT[(m/Sdim)*N + n][Sdim] (used to produce kvT).
__global__ void gemm_wmma(const bf16_t* __restrict__ A,
                          const bf16_t* __restrict__ BT,
                          bf16_t* __restrict__ Cb,
                          float*  __restrict__ Cf,
                          bf16_t* __restrict__ CT,
                          int M, int N, int K, int Sdim) {
    const int lane = threadIdx.x;
    const int half = lane >> 4;
    const int lm   = lane & 15;
    const int ko   = half * 8;
    const int mbase = blockIdx.y * 32;
    const int nbase = blockIdx.x * 64;

    v8f c[2][4] = {};

    const bf16_t* a0p = A + (size_t)(mbase + lm) * K + ko;
    const bf16_t* a1p = A + (size_t)(mbase + 16 + lm) * K + ko;
    const bf16_t* bp0 = BT + (size_t)(nbase +  0 + lm) * K + ko;
    const bf16_t* bp1 = BT + (size_t)(nbase + 16 + lm) * K + ko;
    const bf16_t* bp2 = BT + (size_t)(nbase + 32 + lm) * K + ko;
    const bf16_t* bp3 = BT + (size_t)(nbase + 48 + lm) * K + ko;

    for (int kb = 0; kb < K; kb += 32) {
        v16bf a0 = ld_op(a0p + kb);
        v16bf a1 = ld_op(a1p + kb);
        v16bf b0 = ld_op(bp0 + kb);
        v16bf b1 = ld_op(bp1 + kb);
        v16bf b2 = ld_op(bp2 + kb);
        v16bf b3 = ld_op(bp3 + kb);
        c[0][0] = wmma_bf16(a0, b0, c[0][0]);
        c[0][1] = wmma_bf16(a0, b1, c[0][1]);
        c[0][2] = wmma_bf16(a0, b2, c[0][2]);
        c[0][3] = wmma_bf16(a0, b3, c[0][3]);
        c[1][0] = wmma_bf16(a1, b0, c[1][0]);
        c[1][1] = wmma_bf16(a1, b1, c[1][1]);
        c[1][2] = wmma_bf16(a1, b2, c[1][2]);
        c[1][3] = wmma_bf16(a1, b3, c[1][3]);
    }

    #pragma unroll
    for (int i = 0; i < 2; i++) {
        #pragma unroll
        for (int j = 0; j < 4; j++) {
            const int n     = nbase + j * 16 + lm;
            const int mrow0 = mbase + i * 16 + half * 8;  // 8 consecutive rows
            if (Cb) {
                #pragma unroll
                for (int r = 0; r < 8; r++)
                    Cb[(size_t)(mrow0 + r) * N + n] = (bf16_t)c[i][j][r];
            }
            if (Cf) {
                #pragma unroll
                for (int r = 0; r < 8; r++)
                    Cf[(size_t)(mrow0 + r) * N + n] = c[i][j][r];
            }
            if (CT) {
                // transposed store: 8 consecutive rows pack into one 16B store
                const int bidx = mrow0 / Sdim;
                const int srow = mrow0 % Sdim;
                v8bf t;
                #pragma unroll
                for (int r = 0; r < 8; r++) t[r] = (bf16_t)c[i][j][r];
                *(v8bf*)(CT + ((size_t)bidx * N + n) * Sdim + srow) = t;
            }
        }
    }
}

// ---------------- fused flash attention ----------------
// One wave per (b, h, 16-row q tile); streams 32-key blocks with online
// softmax. q/kv row-major bf16; V read from kvT [B][H][Dh][S].
#define PST 40   // padded LDS row stride (bf16 elems); 80B rows stay 16B-aligned

__global__ void mla_attn_kernel(const bf16_t* __restrict__ qb,
                                const bf16_t* __restrict__ kvb,
                                const bf16_t* __restrict__ kvT,
                                bf16_t* __restrict__ ob) {
    __shared__ alignas(16) bf16_t pbuf[16 * PST];

    const int lane = threadIdx.x;
    const int half = lane >> 4;
    const int lm   = lane & 15;
    const int ko   = half * 8;
    const int qt   = blockIdx.x;            // q tile: 0..127
    const int bh   = blockIdx.y;            // 0..31
    const int b    = bh >> 4;
    const int h    = bh & 15;
    const int qrow = b * SEQ + qt * 16;
    const float scale = 0.08838834764831845f;  // 1/sqrt(128)

    // Q tile as four chained A operands (d = 4 x 32)
    v16bf aq[4];
    {
        const bf16_t* qp = qb + (size_t)(qrow + lm) * D_MODEL + h * D_HEAD + ko;
        #pragma unroll
        for (int c4 = 0; c4 < 4; c4++) aq[c4] = ld_op(qp + c4 * 32);
    }

    v8f o[8] = {};
    float mr[8], lr[8];
    #pragma unroll
    for (int r = 0; r < 8; r++) { mr[r] = -3.0e30f; lr[r] = 0.0f; }

    const bf16_t* kbase = kvb + (size_t)b * SEQ * D_MODEL + h * D_HEAD + ko;
    const bf16_t* vtb   = kvT + (size_t)(b * N_HEADS + h) * D_HEAD * SEQ;

    for (int kb = 0; kb < SEQ; kb += 32) {
        // scores: S = Q @ K^T, two 16(q)x16(key) tiles
        v8f s0 = {}, s1 = {};
        const bf16_t* kp0 = kbase + (size_t)(kb + lm) * D_MODEL;
        const bf16_t* kp1 = kp0 + (size_t)16 * D_MODEL;
        #pragma unroll
        for (int c4 = 0; c4 < 4; c4++) {
            v16bf bk0 = ld_op(kp0 + c4 * 32);
            v16bf bk1 = ld_op(kp1 + c4 * 32);
            s0 = wmma_bf16(aq[c4], bk0, s0);
            s1 = wmma_bf16(aq[c4], bk1, s1);
        }

        // online softmax per row; lane's element r belongs to row (r+8*half),
        // shfl masks 1/2/4/8 reduce over n=0..15 within each 16-lane half.
        #pragma unroll
        for (int r = 0; r < 8; r++) {
            float mx = fmaxf(s0[r], s1[r]);
            mx = fmaxf(mx, __shfl_xor(mx, 1, 32));
            mx = fmaxf(mx, __shfl_xor(mx, 2, 32));
            mx = fmaxf(mx, __shfl_xor(mx, 4, 32));
            mx = fmaxf(mx, __shfl_xor(mx, 8, 32));
            float mn = fmaxf(mr[r], mx);
            float al = __expf((mr[r] - mn) * scale);
            float p0 = __expf((s0[r] - mn) * scale);
            float p1 = __expf((s1[r] - mn) * scale);
            float sm = p0 + p1;
            sm += __shfl_xor(sm, 1, 32);
            sm += __shfl_xor(sm, 2, 32);
            sm += __shfl_xor(sm, 4, 32);
            sm += __shfl_xor(sm, 8, 32);
            lr[r] = lr[r] * al + sm;
            mr[r] = mn;
            // stage P (D layout) into LDS for A-layout reload
            pbuf[(r + 8 * half) * PST + lm]      = (bf16_t)p0;
            pbuf[(r + 8 * half) * PST + 16 + lm] = (bf16_t)p1;
            #pragma unroll
            for (int dt = 0; dt < 8; dt++) o[dt][r] *= al;
        }
        __syncthreads();
        v16bf ap = ld_op(&pbuf[lm * PST + ko]);   // P as A operand (16x32)
        __syncthreads();

        // O += P @ V ; V B-operand from kvT (contiguous along keys)
        #pragma unroll
        for (int dt = 0; dt < 8; dt++) {
            const bf16_t* vp = vtb + (size_t)(dt * 16 + lm) * SEQ + kb + ko;
            v16bf bv = ld_op(vp);
            o[dt] = wmma_bf16(ap, bv, o[dt]);
        }
    }

    // normalize and store attention output row-major [token, h*128+d]
    float inv[8];
    #pragma unroll
    for (int r = 0; r < 8; r++) inv[r] = 1.0f / lr[r];
    #pragma unroll
    for (int dt = 0; dt < 8; dt++) {
        #pragma unroll
        for (int r = 0; r < 8; r++) {
            int m = qrow + r + 8 * half;
            ob[(size_t)m * D_MODEL + h * D_HEAD + dt * 16 + lm] =
                (bf16_t)(o[dt][r] * inv[r]);
        }
    }
}

// ---------------- host launcher ----------------
extern "C" void kernel_launch(void* const* d_in, const int* in_sizes, int n_in,
                              void* d_out, int out_size, void* d_ws, size_t ws_size,
                              hipStream_t stream) {
    (void)in_sizes; (void)n_in; (void)out_size; (void)ws_size;
    const float* x  = (const float*)d_in[0];
    const float* Wq = (const float*)d_in[1];
    const float* Wd = (const float*)d_in[2];
    const float* Wu = (const float*)d_in[3];
    const float* Wo = (const float*)d_in[4];

    char* ws = (char*)d_ws;
    size_t off = 0;
    auto alloc = [&](size_t bytes) -> void* {
        void* p = ws + off;
        off += (bytes + 255) & ~(size_t)255;
        return p;
    };
    bf16_t* xb   = (bf16_t*)alloc((size_t)MTOT * D_MODEL * 2);
    bf16_t* WqT  = (bf16_t*)alloc((size_t)D_MODEL * D_MODEL * 2);   // [2048][2048]
    bf16_t* WdT  = (bf16_t*)alloc((size_t)D_LATENT * D_MODEL * 2);  // [512][2048]
    bf16_t* WuT  = (bf16_t*)alloc((size_t)D_MODEL * D_LATENT * 2);  // [2048][512]
    bf16_t* WoT  = (bf16_t*)alloc((size_t)D_MODEL * D_MODEL * 2);
    bf16_t* qbuf = (bf16_t*)alloc((size_t)MTOT * D_MODEL * 2);
    bf16_t* lat  = (bf16_t*)alloc((size_t)MTOT * D_LATENT * 2);
    bf16_t* kvb  = (bf16_t*)alloc((size_t)MTOT * D_MODEL * 2);
    bf16_t* kvT  = (bf16_t*)alloc((size_t)MTOT * D_MODEL * 2);      // [B][H][Dh][S]
    bf16_t* obuf = (bf16_t*)alloc((size_t)MTOT * D_MODEL * 2);

    { int n = MTOT * D_MODEL;
      cvt_kernel<<<n / 256, 256, 0, stream>>>(x, xb, n); }
    { int n = D_MODEL * D_MODEL;
      transpose_cvt_kernel<<<n / 256, 256, 0, stream>>>(Wq, WqT, D_MODEL, D_MODEL); }
    { int n = D_MODEL * D_LATENT;
      transpose_cvt_kernel<<<n / 256, 256, 0, stream>>>(Wd, WdT, D_MODEL, D_LATENT); }
    { int n = D_LATENT * D_MODEL;
      transpose_cvt_kernel<<<n / 256, 256, 0, stream>>>(Wu, WuT, D_LATENT, D_MODEL); }
    { int n = D_MODEL * D_MODEL;
      transpose_cvt_kernel<<<n / 256, 256, 0, stream>>>(Wo, WoT, D_MODEL, D_MODEL); }

    dim3 blk(32, 1, 1);
    // q = x @ Wq          (bf16 out)
    gemm_wmma<<<dim3(D_MODEL / 64, MTOT / 32), blk, 0, stream>>>(
        xb, WqT, qbuf, nullptr, nullptr, MTOT, D_MODEL, D_MODEL, SEQ);
    // latent = x @ Wdown  (bf16 out)
    gemm_wmma<<<dim3(D_LATENT / 64, MTOT / 32), blk, 0, stream>>>(
        xb, WdT, lat, nullptr, nullptr, MTOT, D_LATENT, D_MODEL, SEQ);
    // kv = latent @ Wup   (bf16 row-major + transposed kvT)
    gemm_wmma<<<dim3(D_MODEL / 64, MTOT / 32), blk, 0, stream>>>(
        lat, WuT, kvb, nullptr, kvT, MTOT, D_MODEL, D_LATENT, SEQ);
    // fused attention
    mla_attn_kernel<<<dim3(SEQ / 16, BATCH * N_HEADS), blk, 0, stream>>>(
        qbuf, kvb, kvT, obuf);
    // final = attn_out @ Wo (fp32 out)
    gemm_wmma<<<dim3(D_MODEL / 64, MTOT / 32), blk, 0, stream>>>(
        obuf, WoT, nullptr, (float*)d_out, nullptr, MTOT, D_MODEL, D_MODEL, SEQ);
}
